// RawGAT_42889543418176
// MI455X (gfx1250) — compile-verified, matrix operand order checked by reading, not run on hardware
//
#include <hip/hip_runtime.h>
#include <hip/hip_bf16.h>

typedef __attribute__((ext_vector_type(16))) _Float16 v16h;
typedef __attribute__((ext_vector_type(8)))  _Float16 v8h;
typedef __attribute__((ext_vector_type(8)))  float    v8f;

constexpr int CB  = 16;
constexpr int CT  = 12;
constexpr int CT1 = 13;
constexpr int CN  = 400;
constexpr int CNP = 416;              // padded attention length (13 chunks of 32)
constexpr int CR  = CB * CT * CN;     // 76800 rows of x
constexpr int CR1 = CB * CT1 * CN;    // 83200 rows of covariate features
constexpr int LDT = 160;              // padded row stride for the 135-col operand

// ---------------------------------------------------------------------------
// small utility kernels
// ---------------------------------------------------------------------------
__global__ void k_zero(float* p, int n) {
    int i = blockIdx.x * blockDim.x + threadIdx.x;
    if (i < n) p[i] = 0.f;
}

// pack a weight matrix into transposed f16 [N][Kpad] (zero-padded K tail).
// headmode=0: src is row-major K x N.  headmode=1: src is (heads, K, 32) and
// logical column n = h*32 + d maps to src[((n>>5)*K + k)*32 + (n&31)].
__global__ void k_pack_bt(const float* __restrict__ src, _Float16* __restrict__ dst,
                          int K, int N, int Kpad, int headmode) {
    int i = blockIdx.x * blockDim.x + threadIdx.x;
    if (i >= N * Kpad) return;
    int n = i / Kpad, k = i % Kpad;
    float v = 0.f;
    if (k < K)
        v = headmode ? src[((size_t)(n >> 5) * K + k) * 32 + (n & 31)]
                     : src[(size_t)k * N + n];
    dst[i] = (_Float16)v;
}

// ---------------------------------------------------------------------------
// feature embedding  (FM first/second order + deep tower pre-BN)
// ---------------------------------------------------------------------------
struct EmbedArgs {
    const float* fm1[7];
    const float* fm2[7];
    const float* Wd;
    const float* bd;
};

__global__ __launch_bounds__(64) void k_embed_pass1(
    const float* __restrict__ cov, EmbedArgs ea,
    float* __restrict__ first7, float* __restrict__ second,
    float* __restrict__ deep) {
    const int FS[7] = {2, 1, 1, 400, 7, 24, 2};
    int rr = blockIdx.x;           // row in [0, CR1)
    int d  = threadIdx.x;          // 0..63
    __shared__ float sg[7][64];
    __shared__ float sfirst[7];

    float f2v[7];
#pragma unroll
    for (int i = 0; i < 7; ++i) {
        float fv = cov[(size_t)rr * 7 + i];
        int   ii;
        float v;
        if (i == 1 || i == 2) { ii = 0; v = fv; }
        else {
            int r = (int)rintf(fv);
            ii = r < 0 ? 0 : (r > FS[i] - 1 ? FS[i] - 1 : r);
            v  = 1.f;
        }
        float g = ea.fm2[i][(size_t)ii * 64 + d] * v;
        sg[i][d] = g;
        f2v[i]   = g;
        if (d == 0) sfirst[i] = ea.fm1[i][ii] * v;
    }
    float s = 0.f, sq = 0.f;
#pragma unroll
    for (int i = 0; i < 7; ++i) { s += f2v[i]; sq += f2v[i] * f2v[i]; }
    second[(size_t)rr * 64 + d] = 0.5f * (s * s - sq);

    __syncthreads();
    const float* sgf = &sg[0][0];
    float acc = ea.bd[d];
#pragma unroll 4
    for (int k = 0; k < 448; ++k) acc = fmaf(sgf[k], ea.Wd[(size_t)k * 64 + d], acc);
    deep[(size_t)rr * 64 + d] = acc;
    if (d < 7) first7[(size_t)rr * 7 + d] = sfirst[d];
}

__global__ __launch_bounds__(256) void k_colstats(const float* __restrict__ deep,
                                                  float* __restrict__ stats, int rows) {
    __shared__ float s1[256], s2[256];
    int col = threadIdx.x & 63, rq = threadIdx.x >> 6;
    float a = 0.f, b = 0.f;
    for (int k = 0; k < 64; ++k) {
        int r = blockIdx.x * 256 + rq + 4 * k;
        if (r < rows) {
            float v = deep[(size_t)r * 64 + col];
            a += v; b += v * v;
        }
    }
    s1[threadIdx.x] = a; s2[threadIdx.x] = b;
    __syncthreads();
    if (threadIdx.x < 64) {
        float ta = 0.f, tb = 0.f;
        for (int q = 0; q < 4; ++q) { ta += s1[q * 64 + threadIdx.x]; tb += s2[q * 64 + threadIdx.x]; }
        atomicAdd(&stats[threadIdx.x], ta);
        atomicAdd(&stats[64 + threadIdx.x], tb);
    }
}

// assemble total = [first(7) | second(64) | relu(BN(deep))(64)] into lda=160 rows
__global__ __launch_bounds__(160) void k_embed_pass2(
    const float* __restrict__ first7, const float* __restrict__ second,
    const float* __restrict__ deep, const float* __restrict__ stats,
    const float* __restrict__ bng, const float* __restrict__ bnb,
    float* __restrict__ tot, int rows) {
    int t = threadIdx.x;
    int rr = blockIdx.x;
    float v = 0.f;
    if (t < 7)       v = first7[(size_t)rr * 7 + t];
    else if (t < 71) v = second[(size_t)rr * 64 + (t - 7)];
    else if (t < 135) {
        int d = t - 71;
        float cnt = (float)rows;
        float mu  = stats[d] / cnt;
        float var = stats[64 + d] / cnt - mu * mu;
        float dv  = deep[(size_t)rr * 64 + d];
        v = (dv - mu) * rsqrtf(var + 1e-5f) * bng[d] + bnb[d];
        v = fmaxf(v, 0.f);
    }
    tot[(size_t)rr * LDT + t] = v;   // cols 135..159 written 0 by threads 135..159
}

// ---------------------------------------------------------------------------
// convs over T (kernel 3, pad 1)
// ---------------------------------------------------------------------------
__global__ __launch_bounds__(64) void k_conv1(const float* __restrict__ enc,
                                              const float* __restrict__ w,
                                              const float* __restrict__ bias,
                                              float* __restrict__ out) {
    int row = blockIdx.x, d = threadIdx.x;
    int n = row % CN, t = (row / CN) % CT, b = row / (CN * CT);
    __shared__ float sx[3][2];
    if (threadIdx.x < 6) {
        int kt = threadIdx.x >> 1, c = threadIdx.x & 1;
        int tt = t + kt - 1;
        sx[kt][c] = (tt >= 0 && tt < CT) ? enc[(((size_t)b * CT + tt) * CN + n) * 2 + c] : 0.f;
    }
    __syncthreads();
    float acc = bias[d];
#pragma unroll
    for (int c = 0; c < 2; ++c)
#pragma unroll
        for (int kt = 0; kt < 3; ++kt)
            acc = fmaf(sx[kt][c], w[(d * 2 + c) * 3 + kt], acc);
    out[(size_t)row * 64 + d] = fmaxf(acc, 0.f);
}

__global__ __launch_bounds__(64) void k_conv2(const float* __restrict__ x1,
                                              const float* __restrict__ w,
                                              const float* __restrict__ bias,
                                              float* __restrict__ out) {
    int row = blockIdx.x, d = threadIdx.x;
    int n = row % CN, t = (row / CN) % CT, b = row / (CN * CT);
    __shared__ float sx[3][64];
    for (int idx = threadIdx.x; idx < 192; idx += 64) {
        int kt = idx >> 6, c = idx & 63;
        int tt = t + kt - 1;
        sx[kt][c] = (tt >= 0 && tt < CT) ? x1[(((size_t)b * CT + tt) * CN + n) * 64 + c] : 0.f;
    }
    __syncthreads();
    float acc = bias[d];
#pragma unroll 1
    for (int kt = 0; kt < 3; ++kt)
#pragma unroll 8
        for (int c = 0; c < 64; ++c)
            acc = fmaf(sx[kt][c], w[(d * 64 + c) * 3 + kt], acc);
    out[(size_t)row * 64 + d] = acc;
}

// ---------------------------------------------------------------------------
// WMMA GEMM: C[M x N] = A[M x Kpad(f32)] * Bp[N x Kpad(f16, transposed)]
// Requirements (all call sites satisfy): M % 64 == 0, N % 32 == 0,
// Kpad % 32 == 0, lda >= Kpad (zero/finite pad), Bp zero-padded.
// No LDS, no barriers: A frag = 4x b128 global loads + cvt; B frag = 2x b128
// global loads (weights stay hot in cache). Each wave computes a 16x32 tile.
// ---------------------------------------------------------------------------
__global__ __launch_bounds__(128) void k_gemm_wmma(
    const float* __restrict__ A, int Kpad, int lda,
    const _Float16* __restrict__ Bp,
    const float* __restrict__ bias,
    const float* __restrict__ resid, int ldr,
    float* __restrict__ C, int ldc, int c_off, int mode /*0 none,1 relu,2 sigmoid*/) {
    int wave = threadIdx.x >> 5, lane = threadIdx.x & 31;
    int m0 = blockIdx.x * 64, n0 = blockIdx.y * 32;
    int row = lane & 15;
    int kb  = (lane < 16) ? 0 : 8;
    const float*    arow = A  + (size_t)(m0 + wave * 16 + row) * lda;
    const _Float16* br0  = Bp + (size_t)(n0 + row) * Kpad + kb;
    const _Float16* br1  = Bp + (size_t)(n0 + 16 + row) * Kpad + kb;
    v8f acc0 = {0.f, 0.f, 0.f, 0.f, 0.f, 0.f, 0.f, 0.f};
    v8f acc1 = {0.f, 0.f, 0.f, 0.f, 0.f, 0.f, 0.f, 0.f};

    for (int k0 = 0; k0 < Kpad; k0 += 32) {
        const float* ap = arow + k0 + kb;
        __builtin_prefetch(ap + 32, 0, 1);          // next K chunk -> global_prefetch
        float4 a0 = *(const float4*)(ap);
        float4 a1 = *(const float4*)(ap + 4);
        float4 a2 = *(const float4*)(ap + 16);
        float4 a3 = *(const float4*)(ap + 20);
        v16h af;
        af[0]  = (_Float16)a0.x; af[1]  = (_Float16)a0.y; af[2]  = (_Float16)a0.z; af[3]  = (_Float16)a0.w;
        af[4]  = (_Float16)a1.x; af[5]  = (_Float16)a1.y; af[6]  = (_Float16)a1.z; af[7]  = (_Float16)a1.w;
        af[8]  = (_Float16)a2.x; af[9]  = (_Float16)a2.y; af[10] = (_Float16)a2.z; af[11] = (_Float16)a2.w;
        af[12] = (_Float16)a3.x; af[13] = (_Float16)a3.y; af[14] = (_Float16)a3.z; af[15] = (_Float16)a3.w;

        v8h b00 = *(const v8h*)(br0 + k0), b01 = *(const v8h*)(br0 + k0 + 16);
        v8h b10 = *(const v8h*)(br1 + k0), b11 = *(const v8h*)(br1 + k0 + 16);
        v16h bf0, bf1;
#pragma unroll
        for (int u = 0; u < 8; ++u) {
            bf0[u] = b00[u]; bf0[8 + u] = b01[u];
            bf1[u] = b10[u]; bf1[8 + u] = b11[u];
        }
        acc0 = __builtin_amdgcn_wmma_f32_16x16x32_f16(false, af, false, bf0,
                                                      (short)0, acc0, false, false);
        acc1 = __builtin_amdgcn_wmma_f32_16x16x32_f16(false, af, false, bf1,
                                                      (short)0, acc1, false, false);
    }

    int gn0 = n0 + (lane & 15);
    int gn1 = gn0 + 16;
#pragma unroll
    for (int i = 0; i < 8; ++i) {
        int gm = m0 + wave * 16 + ((lane < 16) ? i : 8 + i);
        float v0 = acc0[i], v1 = acc1[i];
        if (bias)  { v0 += bias[gn0]; v1 += bias[gn1]; }
        if (resid) { v0 += resid[(size_t)gm * ldr + gn0]; v1 += resid[(size_t)gm * ldr + gn1]; }
        if (mode == 1)      { v0 = fmaxf(v0, 0.f); v1 = fmaxf(v1, 0.f); }
        else if (mode == 2) { v0 = 1.f / (1.f + __expf(-v0)); v1 = 1.f / (1.f + __expf(-v1)); }
        C[(size_t)gm * ldc + c_off + gn0] = v0;
        C[(size_t)gm * ldc + c_off + gn1] = v1;
    }
}

// ---------------------------------------------------------------------------
// spatial GAT: per-(bt,head) precompute es/ed and transposed f16 value matrix
// ---------------------------------------------------------------------------
__global__ __launch_bounds__(128) void k_esed(
    const float* __restrict__ hsp, const float* __restrict__ a_src,
    const float* __restrict__ a_dst, float* __restrict__ es,
    float* __restrict__ ed, _Float16* __restrict__ hT, int heads) {
    int p = blockIdx.x;   // bt*heads + head
    int j = blockIdx.y * blockDim.x + threadIdx.x;
    if (j >= CNP) return;
    int bt = p / heads, head = p % heads;
    if (j < CN) {
        const float* hrow = hsp + ((size_t)(bt * CN + j)) * (heads * 32) + head * 32;
        float s1 = 0.f, s2 = 0.f;
#pragma unroll
        for (int d = 0; d < 32; ++d) {
            float hv = hrow[d];
            s1 = fmaf(hv, a_src[head * 32 + d], s1);
            s2 = fmaf(hv, a_dst[head * 32 + d], s2);
            hT[((size_t)p * 32 + d) * CNP + j] = (_Float16)hv;
        }
        es[(size_t)p * CNP + j] = s1;
        ed[(size_t)p * CNP + j] = s2;
    } else {
        es[(size_t)p * CNP + j] = -1e30f;
        ed[(size_t)p * CNP + j] = -1e30f;
#pragma unroll
        for (int d = 0; d < 32; ++d) hT[((size_t)p * 32 + d) * CNP + j] = (_Float16)0.f;
    }
}

// flash-attention style GAT: one wave per 16-row tile, online softmax,
// p @ h via two v_wmma_f32_16x16x32_f16 per 32-col chunk.
__global__ __launch_bounds__(32) void k_gat_flash(
    const _Float16* __restrict__ hT, const float* __restrict__ es,
    const float* __restrict__ ed, float* __restrict__ outb, int heads) {
    int p  = blockIdx.x;
    int i0 = blockIdx.y * 16;
    int bt = p / heads, head = p % heads;
    int lane = threadIdx.x;
    const _Float16* hp  = hT + (size_t)p * 32 * CNP;
    const float*    edp = ed + (size_t)p * CNP;
    int row = lane & 15;
    int kb  = (lane < 16) ? 0 : 8;
    float es_r = es[(size_t)p * CNP + i0 + row];

    float m = -1e30f, l = 0.f;
    v8f acc0 = {0.f, 0.f, 0.f, 0.f, 0.f, 0.f, 0.f, 0.f};
    v8f acc1 = {0.f, 0.f, 0.f, 0.f, 0.f, 0.f, 0.f, 0.f};

    for (int j0 = 0; j0 < CNP; j0 += 32) {
        const float* ep = edp + j0 + kb;
        __builtin_prefetch(ep + 32, 0, 1);
        float4 e0 = *(const float4*)(ep);
        float4 e1 = *(const float4*)(ep + 4);
        float4 e2 = *(const float4*)(ep + 16);
        float4 e3 = *(const float4*)(ep + 20);
        float e[16] = {e0.x, e0.y, e0.z, e0.w, e1.x, e1.y, e1.z, e1.w,
                       e2.x, e2.y, e2.z, e2.w, e3.x, e3.y, e3.z, e3.w};
        float mloc = -1e30f;
#pragma unroll
        for (int u = 0; u < 16; ++u) {
            float v = es_r + e[u];
            v = v > 0.f ? v : 0.2f * v;    // leaky_relu 0.2
            e[u] = v;
            mloc = fmaxf(mloc, v);
        }
        mloc = fmaxf(mloc, __shfl_xor(mloc, 16, 32));
        float mnew  = fmaxf(m, mloc);
        float scale = __expf(m - mnew);
        m = mnew;

        v16h af;
        float rsum = 0.f;
#pragma unroll
        for (int u = 0; u < 8; ++u) {
            float p1 = __expf(e[u] - mnew);
            float p2 = __expf(e[8 + u] - mnew);
            rsum += p1 + p2;
            af[u]     = (_Float16)p1;   // k = kb + u
            af[8 + u] = (_Float16)p2;   // k = kb + 16 + u
        }
        rsum += __shfl_xor(rsum, 16, 32);
        l = l * scale + rsum;

#pragma unroll
        for (int i = 0; i < 8; ++i) {
            float si = __shfl(scale, i + ((lane & 16) >> 1), 32);
            acc0[i] *= si;
            acc1[i] *= si;
        }

        const _Float16* hb0 = hp + (size_t)(lane & 15) * CNP + j0 + kb;
        const _Float16* hb1 = hp + (size_t)(16 + (lane & 15)) * CNP + j0 + kb;
        v8h lo0 = *(const v8h*)hb0,  hi0 = *(const v8h*)(hb0 + 16);
        v8h lo1 = *(const v8h*)hb1,  hi1 = *(const v8h*)(hb1 + 16);
        v16h bf0, bf1;
#pragma unroll
        for (int u = 0; u < 8; ++u) {
            bf0[u] = lo0[u]; bf0[8 + u] = hi0[u];
            bf1[u] = lo1[u]; bf1[8 + u] = hi1[u];
        }
        acc0 = __builtin_amdgcn_wmma_f32_16x16x32_f16(false, af, false, bf0,
                                                      (short)0, acc0, false, false);
        acc1 = __builtin_amdgcn_wmma_f32_16x16x32_f16(false, af, false, bf1,
                                                      (short)0, acc1, false, false);
    }

#pragma unroll
    for (int i = 0; i < 8; ++i) {
        int Mi = (lane < 16) ? i : 8 + i;
        float li  = __shfl(l, Mi, 32);
        float inv = 1.f / li;
        size_t gr = (size_t)(bt * CN + i0 + Mi) * (heads * 32) + head * 32;
        outb[gr + (lane & 15)]      = acc0[i] * inv;
        outb[gr + 16 + (lane & 15)] = acc1[i] * inv;
    }
}

// ---------------------------------------------------------------------------
// temporal GAT (attention over only 2 timesteps within each period)
// ---------------------------------------------------------------------------
__global__ __launch_bounds__(128) void k_temporal_attn(
    const float* __restrict__ th, const float* __restrict__ tas,
    const float* __restrict__ tad, float* __restrict__ tatt) {
    int idx = blockIdx.x * blockDim.x + threadIdx.x;
    int tot = CB * CN * 6 * 4;
    if (idx >= tot) return;
    int head = idx & 3, pp = (idx >> 2) % 6, n = (idx / 24) % CN, b = idx / (24 * CN);
    size_t r0 = ((size_t)b * CT + pp * 2) * CN + n;
    size_t r1 = r0 + CN;
    const float* h0 = th + r0 * 128 + head * 32;
    const float* h1 = th + r1 * 128 + head * 32;
    float es0 = 0.f, es1 = 0.f, ed0 = 0.f, ed1 = 0.f;
#pragma unroll
    for (int d = 0; d < 32; ++d) {
        float a = tas[head * 32 + d], c = tad[head * 32 + d];
        es0 = fmaf(h0[d], a, es0); es1 = fmaf(h1[d], a, es1);
        ed0 = fmaf(h0[d], c, ed0); ed1 = fmaf(h1[d], c, ed1);
    }
    auto lrelu = [](float v) { return v > 0.f ? v : 0.2f * v; };
    float e00 = lrelu(es0 + ed0), e01 = lrelu(es0 + ed1);
    float e10 = lrelu(es1 + ed0), e11 = lrelu(es1 + ed1);
    float m0 = fmaxf(e00, e01), m1 = fmaxf(e10, e11);
    float p00 = __expf(e00 - m0), p01 = __expf(e01 - m0);
    float p10 = __expf(e10 - m1), p11 = __expf(e11 - m1);
    float i0 = 1.f / (p00 + p01), i1 = 1.f / (p10 + p11);
    float a00 = p00 * i0, a01 = p01 * i0, a10 = p10 * i1, a11 = p11 * i1;
#pragma unroll
    for (int d = 0; d < 32; ++d) {
        tatt[r0 * 128 + head * 32 + d] = a00 * h0[d] + a01 * h1[d];
        tatt[r1 * 128 + head * 32 + d] = a10 * h0[d] + a11 * h1[d];
    }
}

// gated fusion: x = z*spatial + (1-z)*temporal
__global__ void k_combine(const float* __restrict__ concat, const float* __restrict__ z,
                          float* __restrict__ x) {
    int idx = blockIdx.x * blockDim.x + threadIdx.x;
    if (idx >= CR * 64) return;
    int row = idx >> 6, c = idx & 63;
    float sp = concat[(size_t)row * 128 + c];
    float tp = concat[(size_t)row * 128 + 64 + c];
    float zz = z[idx];
    x[idx] = zz * sp + (1.f - zz) * tp;
}

// ---------------------------------------------------------------------------
// final covariate attention + output projection
// ---------------------------------------------------------------------------
__global__ __launch_bounds__(64) void k_output(
    const float* __restrict__ feats, const float* __restrict__ x,
    const float* __restrict__ Wout, const float* __restrict__ bout,
    float* __restrict__ out) {
    int idx = blockIdx.x * blockDim.x + threadIdx.x;
    if (idx >= CB * CN) return;
    int b = idx / CN, n = idx % CN;
    float tg[64];
#pragma unroll 8
    for (int f = 0; f < 64; ++f)
        tg[f] = feats[(((size_t)b * CT1 + CT) * CN + n) * 64 + f];
    float sc[CT], mx = -1e30f;
    for (int t = 0; t < CT; ++t) {
        const float* sr = feats + (((size_t)b * CT1 + t) * CN + n) * 64;
        float s = 0.f;
#pragma unroll 8
        for (int f = 0; f < 64; ++f) s = fmaf(sr[f], tg[f], s);
        sc[t] = s;
        mx = fmaxf(mx, s);
    }
    float den = 0.f;
    for (int t = 0; t < CT; ++t) { sc[t] = __expf(sc[t] - mx); den += sc[t]; }
    float inv = 1.f / den;
    float o[64];
#pragma unroll 8
    for (int f = 0; f < 64; ++f) o[f] = 0.f;
    for (int t = 0; t < CT; ++t) {
        float wgt = sc[t] * inv;
        const float* xr = x + (((size_t)b * CT + t) * CN + n) * 64;
#pragma unroll 8
        for (int f = 0; f < 64; ++f) o[f] = fmaf(xr[f], wgt, o[f]);
    }
#pragma unroll
    for (int c = 0; c < 2; ++c) {
        float r = bout[c];
#pragma unroll 8
        for (int f = 0; f < 64; ++f) r = fmaf(o[f], Wout[f * 2 + c], r);
        out[((size_t)b * CN + n) * 2 + c] = r;
    }
}

// ---------------------------------------------------------------------------
// host orchestration
// ---------------------------------------------------------------------------
extern "C" void kernel_launch(void* const* d_in, const int* in_sizes, int n_in,
                              void* d_out, int out_size, void* d_ws, size_t ws_size,
                              hipStream_t stream) {
    (void)n_in; (void)out_size; (void)ws_size;
    auto F = [&](int i) { return (const float*)d_in[i]; };

    const float *enc, *cov, *fm1[7], *fm2[7], *Wd, *bd, *bng, *bnb, *Wcat, *bcat;
    const float *c1w, *c1b, *c2w, *c2b, *Wout, *bout;
    struct Layer { const float *sWh, *sas, *sad, *sWo, *sbo, *tWh, *tas, *tad, *tWo, *tbo, *Wg, *bg; } ly[2];

    if (in_sizes[0] == CB * CT * CN * 2) {          // dict insertion order
        enc = F(0); cov = F(1);
        for (int i = 0; i < 7; ++i) fm1[i] = F(2 + i);
        for (int i = 0; i < 7; ++i) fm2[i] = F(9 + i);
        Wd = F(16); bd = F(17); bng = F(18); bnb = F(19); Wcat = F(20); bcat = F(21);
        c1w = F(22); c1b = F(23); c2w = F(24); c2b = F(25);
        for (int L = 0; L < 2; ++L) {
            int o = 26 + 12 * L;
            ly[L] = {F(o), F(o + 1), F(o + 2), F(o + 3), F(o + 4), F(o + 5),
                     F(o + 6), F(o + 7), F(o + 8), F(o + 9), F(o + 10), F(o + 11)};
        }
        Wout = F(50); bout = F(51);
    } else {                                        // jax-sorted flatten order
        cov = F(0); enc = F(1);
        Wcat = F(2); Wd = F(3); Wout = F(4); bcat = F(5); bd = F(6); bout = F(7);
        bnb = F(8); bng = F(9); c1b = F(10); c1w = F(11); c2b = F(12); c2w = F(13);
        for (int i = 0; i < 7; ++i) fm1[i] = F(14 + i);
        for (int i = 0; i < 7; ++i) fm2[i] = F(21 + i);
        for (int L = 0; L < 2; ++L) {
            int o = 28 + 12 * L;   // Wg,bg,sWh,sWo,sa_dst,sa_src,sbo,tWh,tWo,ta_dst,ta_src,tbo
            ly[L].Wg = F(o);       ly[L].bg  = F(o + 1);
            ly[L].sWh = F(o + 2);  ly[L].sWo = F(o + 3);
            ly[L].sad = F(o + 4);  ly[L].sas = F(o + 5); ly[L].sbo = F(o + 6);
            ly[L].tWh = F(o + 7);  ly[L].tWo = F(o + 8);
            ly[L].tad = F(o + 9);  ly[L].tas = F(o + 10); ly[L].tbo = F(o + 11);
        }
    }

    // ---- workspace layout (bump allocator, 256B aligned) ----
    char* w = (char*)d_ws;
    auto alloc = [&](size_t bytes) { char* p = w; w += (bytes + 255) & ~(size_t)255; return p; };
    float*     bnst   = (float*)alloc(1024);
    float*     first7 = (float*)alloc((size_t)CR1 * 7 * 4);
    float*     second = (float*)alloc((size_t)CR1 * 64 * 4);
    float*     deep   = (float*)alloc((size_t)CR1 * 64 * 4);
    float*     tot135 = (float*)alloc((size_t)CR1 * LDT * 4);
    float*     feats  = (float*)alloc((size_t)CR1 * 64 * 4);
    float*     xbuf   = (float*)alloc((size_t)CR * 64 * 4);
    float*     x1buf  = (float*)alloc((size_t)CR * 64 * 4);
    float*     hsp    = (float*)alloc((size_t)CR * 96 * 4);
    _Float16*  hT     = (_Float16*)alloc((size_t)576 * 32 * CNP * 2);
    float*     esb    = (float*)alloc((size_t)576 * CNP * 4);
    float*     edb    = (float*)alloc((size_t)576 * CNP * 4);
    float*     attsp  = (float*)alloc((size_t)CR * 96 * 4);
    float*     cats   = (float*)alloc((size_t)CR * 128 * 4);
    float*     thbuf  = (float*)alloc((size_t)CR * 128 * 4);
    float*     tatt   = (float*)alloc((size_t)CR * 128 * 4);
    float*     zbuf   = (float*)alloc((size_t)CR * 64 * 4);
    // packed f16 weights (transposed [N][Kpad])
    _Float16*  pbWcat = (_Float16*)alloc((size_t)64  * 160 * 2);
    _Float16*  pbWh   = (_Float16*)alloc((size_t)96  * 64  * 2);
    _Float16*  pbWo   = (_Float16*)alloc((size_t)64  * 96  * 2);
    _Float16*  pbTWh  = (_Float16*)alloc((size_t)128 * 64  * 2);
    _Float16*  pbTWo  = (_Float16*)alloc((size_t)64  * 128 * 2);
    _Float16*  pbWg   = (_Float16*)alloc((size_t)64  * 128 * 2);

    // ---- feature embedding ----
    k_zero<<<1, 256, 0, stream>>>(bnst, 256);
    EmbedArgs ea;
    for (int i = 0; i < 7; ++i) { ea.fm1[i] = fm1[i]; ea.fm2[i] = fm2[i]; }
    ea.Wd = Wd; ea.bd = bd;
    k_embed_pass1<<<CR1, 64, 0, stream>>>(cov, ea, first7, second, deep);
    k_colstats<<<(CR1 + 255) / 256, 256, 0, stream>>>(deep, bnst, CR1);
    k_embed_pass2<<<CR1, 160, 0, stream>>>(first7, second, deep, bnst, bng, bnb, tot135, CR1);
    k_pack_bt<<<(64 * 160 + 127) / 128, 128, 0, stream>>>(Wcat, pbWcat, 135, 64, 160, 0);
    k_gemm_wmma<<<dim3(CR1 / 64, 2), 128, 0, stream>>>(
        tot135, 160, LDT, pbWcat, bcat, nullptr, 0, feats, 64, 0, 0);

    // ---- input convs ----
    k_conv1<<<CR, 64, 0, stream>>>(enc, c1w, c1b, x1buf);
    k_conv2<<<CR, 64, 0, stream>>>(x1buf, c2w, c2b, xbuf);

    // ---- 2 GAT layers ----
    for (int L = 0; L < 2; ++L) {
        const Layer& lp = ly[L];
        // spatial: h = x @ sWh (64 -> 96)
        k_pack_bt<<<(96 * 64 + 127) / 128, 128, 0, stream>>>(lp.sWh, pbWh, 64, 96, 64, 1);
        k_gemm_wmma<<<dim3(CR / 64, 3), 128, 0, stream>>>(
            xbuf, 64, 64, pbWh, nullptr, nullptr, 0, hsp, 96, 0, 0);
        k_esed<<<dim3(576, (CNP + 127) / 128), 128, 0, stream>>>(hsp, lp.sas, lp.sad, esb, edb, hT, 3);
        k_gat_flash<<<dim3(576, CN / 16), 32, 0, stream>>>(hT, esb, edb, attsp, 3);
        // spatial = x + att @ sWo + sbo  -> concat[:, 0:64]
        k_pack_bt<<<(64 * 96 + 127) / 128, 128, 0, stream>>>(lp.sWo, pbWo, 96, 64, 96, 0);
        k_gemm_wmma<<<dim3(CR / 64, 2), 128, 0, stream>>>(
            attsp, 96, 96, pbWo, lp.sbo, xbuf, 64, cats, 128, 0, 0);
        // temporal: t_h = x @ tWh (64 -> 128)
        k_pack_bt<<<(128 * 64 + 127) / 128, 128, 0, stream>>>(lp.tWh, pbTWh, 64, 128, 64, 1);
        k_gemm_wmma<<<dim3(CR / 64, 4), 128, 0, stream>>>(
            xbuf, 64, 64, pbTWh, nullptr, nullptr, 0, thbuf, 128, 0, 0);
        k_temporal_attn<<<(CB * CN * 6 * 4 + 127) / 128, 128, 0, stream>>>(thbuf, lp.tas, lp.tad, tatt);
        // temporal = x + tatt @ tWo + tbo -> concat[:, 64:128]
        k_pack_bt<<<(64 * 128 + 127) / 128, 128, 0, stream>>>(lp.tWo, pbTWo, 128, 64, 128, 0);
        k_gemm_wmma<<<dim3(CR / 64, 2), 128, 0, stream>>>(
            tatt, 128, 128, pbTWo, lp.tbo, xbuf, 64, cats, 128, 64, 0);
        // gate z = sigmoid(concat @ Wg + bg); x = z*sp + (1-z)*tmp
        k_pack_bt<<<(64 * 128 + 127) / 128, 128, 0, stream>>>(lp.Wg, pbWg, 128, 64, 128, 0);
        k_gemm_wmma<<<dim3(CR / 64, 2), 128, 0, stream>>>(
            cats, 128, 128, pbWg, lp.bg, nullptr, 0, zbuf, 64, 0, 2);
        k_combine<<<(CR * 64 + 255) / 256, 256, 0, stream>>>(cats, zbuf, xbuf);
    }

    // ---- final covariate attention + projection ----
    k_output<<<(CB * CN + 63) / 64, 64, 0, stream>>>(feats, xbuf, Wout, bout, (float*)d_out);
}
